// MyGRUCell_55490977464698
// MI455X (gfx1250) — matrix operand order repaired and checked by reading, not compiled, not run on hardware
//
#include <hip/hip_runtime.h>
#include <hip/hip_bf16.h>
#include <stdint.h>

// ---- WMMA fragment / vector types (gfx1250, wave32) ----
typedef __bf16 bf16_t;
typedef bf16_t v16bf __attribute__((ext_vector_type(16)));
typedef bf16_t v2bf  __attribute__((ext_vector_type(2)));
typedef float  v8f   __attribute__((ext_vector_type(8)));
typedef float  f32x2 __attribute__((ext_vector_type(2)));   // native vector: OK for NT builtins

#define BATCH          65536
#define KDIM           256
#define NDIM           256
#define KP             264            // padded LDS row length in bf16 elems (bank-conflict dodge)
#define KP2            (KP / 2)       // same, in u32 units (132; multiple of 4 -> 16B aligned rows)
#define ROWS_PER_BLOCK 128
#define THREADS        256

__device__ __forceinline__ uint32_t f32_to_bf16_rne(float f) {
  uint32_t b = __float_as_uint(f);
  return (b + 0x7FFFu + ((b >> 16) & 1u)) >> 16;
}
__device__ __forceinline__ uint32_t pack2(float lo, float hi) {
#if __has_builtin(__builtin_amdgcn_cvt_pk_bf16_f32)
  v2bf p = __builtin_amdgcn_cvt_pk_bf16_f32(lo, hi);
  return __builtin_bit_cast(uint32_t, p);
#else
  return f32_to_bf16_rne(lo) | (f32_to_bf16_rne(hi) << 16);
#endif
}
__device__ __forceinline__ float fast_sigmoid(float x) {
  return 1.0f / (1.0f + __expf(-x));
}
__device__ __forceinline__ float fast_tanh(float x) {
  float e = __expf(2.0f * x);
  return (e - 1.0f) / (e + 1.0f);
}

union Frag16 { uint32_t u[8]; uint4 q[2]; v16bf v; };
union FragC  { float    f[8]; v8f   v; };

// B-matrix (32x16 KxN bf16): V0-3 lanes0-15 K=0..7 / lanes16-31 K=8..15,
// V4-7 lanes0-15 K=16..23 / lanes16-31 K=24..31. All chunks 16B aligned -> ds_load_b128.
__device__ __forceinline__ Frag16 load_bfrag(const uint32_t* __restrict__ wp, int o) {
  Frag16 f;
  f.q[0] = *(const uint4*)(wp + o);
  f.q[1] = *(const uint4*)(wp + o + 8);
  return f;
}

// A-matrix (16x32 MxK bf16), ISA layout: V0-3 lanes0-15 K=0..7 / lanes16-31 K=8..15,
// V4-7 lanes0-15 K=16..23 / lanes16-31 K=24..31. 16B aligned -> ds_load_b128.
__device__ __forceinline__ Frag16 load_afrag(const uint32_t* __restrict__ rowp, int o) {
  Frag16 f;
  f.q[0] = *(const uint4*)(rowp + o);
  f.q[1] = *(const uint4*)(rowp + o + 8);
  return f;
}

__global__ __launch_bounds__(THREADS, 1)
void gru_fused_wmma(const float* __restrict__ x,
                    const float* __restrict__ h_prev,
                    const float* __restrict__ W_ir, const float* __restrict__ b_ir,
                    const float* __restrict__ W_iz, const float* __restrict__ b_iz,
                    const float* __restrict__ W_in, const float* __restrict__ b_in,
                    const float* __restrict__ W_hz, const float* __restrict__ b_hz,
                    const float* __restrict__ W_hn, const float* __restrict__ b_hn,
                    float* __restrict__ out)
{
  extern __shared__ uint32_t smem[];
  // LDS layout (u32 units): [5 mats x 16 rows x KP2] weights, then x tile, then h tile
  uint32_t* wlds = smem;                               // 5*16*KP2
  uint32_t* xlds = smem + 5 * 16 * KP2;                // 128*KP2
  uint32_t* hlds = xlds + ROWS_PER_BLOCK * KP2;        // 128*KP2

  const int tid  = threadIdx.x;
  const int row0 = blockIdx.x * ROWS_PER_BLOCK;

  // ---- stage x and h_prev tiles ONCE (coalesced b64 loads), f32 -> bf16 ----
  // x is read exactly once per kernel -> non-temporal; h_prev is re-read in the
  // epilogue -> regular (cache it).
#pragma unroll 4
  for (int it = 0; it < 64; ++it) {
    const int idx = tid + it * THREADS;       // f32x2 units, 0..16383
    const int r   = idx >> 7;
    const int kp  = idx & 127;
    const f32x2 xv =
        __builtin_nontemporal_load((const f32x2*)(x + (size_t)(row0 + r) * KDIM) + kp);
    const f32x2 hv = ((const f32x2*)(h_prev + (size_t)(row0 + r) * KDIM))[kp];
    xlds[r * KP2 + kp] = pack2(xv.x, xv.y);
    hlds[r * KP2 + kp] = pack2(hv.x, hv.y);
  }

  const int lane = tid & 31;
  const int wave = tid >> 5;
  const int half = lane >> 4;
  const int mrow = lane & 15;   // A-row / B-col / C-col index

  const uint32_t* xrow = xlds + (wave * 16 + mrow) * KP2;
  const uint32_t* hrow = hlds + (wave * 16 + mrow) * KP2;
  const uint32_t* wcol = wlds + mrow * KP2;   // column mrow of each staged weight tile

  const float* Wmats[5] = { W_hz, W_iz, W_ir, W_in, W_hn };

  // ---- loop over all 16 column blocks; x/h LDS tiles are reused every pass ----
#pragma unroll 1
  for (int ny = 0; ny < NDIM / 16; ++ny) {
    const int n0 = ny * 16;

    __syncthreads();   // previous pass finished reading wlds
    // stage 16-column slices of the 5 weight matrices, f32 -> bf16
#pragma unroll
    for (int mat = 0; mat < 5; ++mat) {
      const float* Wm = Wmats[mat];
      uint32_t* dst = wlds + mat * 16 * KP2;
#pragma unroll
      for (int it = 0; it < 8; ++it) {
        const int idx = tid + it * THREADS;   // f32x2 units, 0..2047
        const int j   = idx >> 7;             // row within 16-row tile
        const int kp  = idx & 127;            // f32x2 index along K
        const f32x2 w2 = ((const f32x2*)(Wm + (size_t)(n0 + j) * KDIM))[kp];
        dst[j * KP2 + kp] = pack2(w2.x, w2.y);
      }
    }
    __syncthreads();   // staged weights visible

    // 5 f32 accumulators, pre-loaded with the per-column biases
    FragC acc_hz, acc_iz, acc_ir, acc_in, acc_hn;
    {
      const float bhz = b_hz[n0 + mrow];
      const float biz = b_iz[n0 + mrow];
      const float bir = b_ir[n0 + mrow];
      const float bnn = b_in[n0 + mrow];
      const float bhn = b_hn[n0 + mrow];
#pragma unroll
      for (int i = 0; i < 8; ++i) {
        acc_hz.f[i] = bhz; acc_iz.f[i] = biz; acc_ir.f[i] = bir;
        acc_in.f[i] = bnn; acc_hn.f[i] = bhn;
      }
    }

    // ---- main loop: K=256 in 8 steps of 32, 5 WMMAs per step ----
#pragma unroll
    for (int kb = 0; kb < KDIM; kb += 32) {
      const int ku = kb >> 1;                          // u32 offset of this k-slice
      const Frag16 xa = load_afrag(xrow, ku + 4 * half);
      const Frag16 ha = load_afrag(hrow, ku + 4 * half);
      const int ob = ku + 4 * half;

      Frag16 bw;
      bw = load_bfrag(wcol + 0 * 16 * KP2, ob);        // W_hz
      acc_hz.v = __builtin_amdgcn_wmma_f32_16x16x32_bf16(
          false, ha.v, false, bw.v, (short)0, acc_hz.v, false, false);
      bw = load_bfrag(wcol + 4 * 16 * KP2, ob);        // W_hn
      acc_hn.v = __builtin_amdgcn_wmma_f32_16x16x32_bf16(
          false, ha.v, false, bw.v, (short)0, acc_hn.v, false, false);
      bw = load_bfrag(wcol + 1 * 16 * KP2, ob);        // W_iz
      acc_iz.v = __builtin_amdgcn_wmma_f32_16x16x32_bf16(
          false, xa.v, false, bw.v, (short)0, acc_iz.v, false, false);
      bw = load_bfrag(wcol + 2 * 16 * KP2, ob);        // W_ir
      acc_ir.v = __builtin_amdgcn_wmma_f32_16x16x32_bf16(
          false, xa.v, false, bw.v, (short)0, acc_ir.v, false, false);
      bw = load_bfrag(wcol + 3 * 16 * KP2, ob);        // W_in
      acc_in.v = __builtin_amdgcn_wmma_f32_16x16x32_bf16(
          false, xa.v, false, bw.v, (short)0, acc_in.v, false, false);
    }

    // ---- fused gate epilogue; C/D layout: VGPR v -> row (v + 8*half), col mrow ----
    const int ncol = n0 + mrow;
#pragma unroll
    for (int v = 0; v < 8; ++v) {
      const int    mg  = row0 + wave * 16 + half * 8 + v;
      const size_t off = (size_t)mg * NDIM + ncol;
      const float  hp  = h_prev[off];
      const float  hz  = acc_hz.f[v];
      const float  z   = fast_sigmoid(acc_iz.f[v] + hz);
      const float  r   = fast_sigmoid(acc_ir.f[v] + hz);   // reference reuses hz here
      const float  g   = fast_tanh(acc_in.f[v] + r * acc_hn.f[v]);
      const float  hn  = (1.0f - z) * g + z * hp;
      __builtin_nontemporal_store(hn, out + off);          // streamed, never re-read
    }
  }
}

extern "C" void kernel_launch(void* const* d_in, const int* in_sizes, int n_in,
                              void* d_out, int out_size, void* d_ws, size_t ws_size,
                              hipStream_t stream) {
  (void)in_sizes; (void)n_in; (void)out_size; (void)d_ws; (void)ws_size;
  const float* x     = (const float*)d_in[0];
  const float* h     = (const float*)d_in[1];
  const float* W_ir  = (const float*)d_in[2];
  const float* b_ir  = (const float*)d_in[3];
  const float* W_iz  = (const float*)d_in[4];
  const float* b_iz  = (const float*)d_in[5];
  const float* W_in  = (const float*)d_in[6];
  const float* b_in  = (const float*)d_in[7];
  const float* W_hz  = (const float*)d_in[8];
  const float* b_hz  = (const float*)d_in[9];
  const float* W_hn  = (const float*)d_in[10];
  const float* b_hn  = (const float*)d_in[11];
  float* out = (float*)d_out;

  dim3 grid(BATCH / ROWS_PER_BLOCK, 1);
  dim3 block(THREADS);
  size_t smem_bytes = (size_t)(5 * 16 * KP2 + 2 * ROWS_PER_BLOCK * KP2) * sizeof(uint32_t);
  gru_fused_wmma<<<grid, block, smem_bytes, stream>>>(
      x, h, W_ir, b_ir, W_iz, b_iz, W_in, b_in, W_hz, b_hz, W_hn, b_hn, out);
}
